// PhaseBindingBlock_84825604096594
// MI455X (gfx1250) — matrix operand order repaired and checked by reading, not compiled, not run on hardware
//
#include <hip/hip_runtime.h>
#include <hip/hip_bf16.h>
#include <math.h>
#include <stdint.h>

#define BB 2
#define LL 2048
#define DD 512
#define KK 32
#define MM (BB * LL)

typedef __attribute__((ext_vector_type(16))) __bf16 v16bf;
typedef __attribute__((ext_vector_type(8)))  __bf16 v8bf;
typedef __attribute__((ext_vector_type(8)))  float  v8f;
typedef __attribute__((ext_vector_type(4)))  int    v4i;

union ABfrag { v16bf v; v8bf h[2]; };

#if defined(__has_builtin)
#  if __has_builtin(__builtin_amdgcn_global_load_async_to_lds_b128)
#    define HAS_ASYNC_LDS 1
#  else
#    define HAS_ASYNC_LDS 0
#  endif
#else
#  define HAS_ASYNC_LDS 0
#endif

typedef __attribute__((address_space(1))) v4i gv4i;   // global int4
typedef __attribute__((address_space(3))) v4i lv4i;   // LDS int4

// 16-byte global -> LDS copy. Async (ASYNCcnt-tracked) when the toolchain
// exposes the gfx1250 builtin, otherwise staged through VGPRs (ds_store).
__device__ __forceinline__ void cp_b128(void* lds, const void* g) {
#if HAS_ASYNC_LDS
  __builtin_amdgcn_global_load_async_to_lds_b128(
      (gv4i*)(uintptr_t)g,
      (lv4i*)(unsigned int)(uintptr_t)lds,
      0, 0);
#else
  *(v8bf*)lds = *(const v8bf*)g;
#endif
}

template<int N>
__device__ __forceinline__ void wait_async() {
#if HAS_ASYNC_LDS
#  if __has_builtin(__builtin_amdgcn_s_wait_asynccnt)
  __builtin_amdgcn_s_wait_asynccnt(N);
#  else
  if (N == 0) asm volatile("s_wait_asynccnt 0x0" ::: "memory");
  else        asm volatile("s_wait_asynccnt 0x3" ::: "memory");
#  endif
#endif
}

__device__ __forceinline__ unsigned short f2bf(float f) {
  unsigned int u = __float_as_uint(f);
  u += 0x7FFFu + ((u >> 16) & 1u);   // round-to-nearest-even
  return (unsigned short)(u >> 16);
}

// ---------------------------------------------------------------- converts
__global__ void convert_bf16_kernel(const float* __restrict__ in,
                                    unsigned short* __restrict__ out, int n) {
  int i = blockIdx.x * blockDim.x + threadIdx.x;
  if (i < n) out[i] = f2bf(in[i]);
}

// ---------------------------------------------------------------- staged GEMM
// C[M,N] = A[M,Kd] @ W[Kd,N] + bias. 256-thread block computes a 128x64 tile,
// double-buffered async copies into LDS feed v_wmma_f32_16x16x32_bf16.
// EP: 0 = f32 store, 1 = tanh -> bf16, 2 = f32 store + residual add.
template<int EP>
__global__ __launch_bounds__(256)
void gemm_wmma_staged_kernel(const unsigned short* __restrict__ A,
                             const unsigned short* __restrict__ W,
                             const float* __restrict__ bias,
                             const float* __restrict__ resid,
                             float* __restrict__ outF,
                             unsigned short* __restrict__ outB,
                             int M, int N, int Kd)
{
  constexpr int BM = 128, BN = 64, BK = 32;
  constexpr int PA = 8, PW = 8;                 // 16B row pad: kill bank conflicts
  __shared__ unsigned short sA[2][BM][BK + PA]; // 2 x 128 x 40 bf16 = 20 KB
  __shared__ unsigned short sW[2][BK][BN + PW]; // 2 x  32 x 72 bf16 =  9 KB

  const int tIdx  = threadIdx.x;
  const int lane  = tIdx & 31;
  const int wv    = tIdx >> 5;                  // 0..7
  const int hi    = lane >> 4;
  const int nBlks = N / BN;
  const int mBase = (int)(blockIdx.x / nBlks) * BM;
  const int nBase = (int)(blockIdx.x % nBlks) * BN;

  auto stage = [&](int buf, int kk) {
    // A tile: 128x32 bf16 = 512 b128 chunks, 2 per thread
    int c = tIdx, row = c >> 2, seg = c & 3;
    cp_b128(&sA[buf][row][seg * 8],
            A + (size_t)(mBase + row) * Kd + kk + seg * 8);
    c = tIdx + 256; row = c >> 2; seg = c & 3;
    cp_b128(&sA[buf][row][seg * 8],
            A + (size_t)(mBase + row) * Kd + kk + seg * 8);
    // W panel: 32x64 bf16 = 256 b128 chunks, 1 per thread
    row = tIdx >> 3; seg = tIdx & 7;
    cp_b128(&sW[buf][row][seg * 8],
            W + (size_t)(kk + row) * N + nBase + seg * 8);
  };

  v8f acc[4] = {};
  const int S = Kd / BK;

  stage(0, 0);
  for (int s = 0; s < S; ++s) {
    if (s + 1 < S) { stage((s + 1) & 1, (s + 1) * BK); wait_async<3>(); }
    else           { wait_async<0>(); }
    __syncthreads();

    const int buf = s & 1;
    // A fragment, 16-bit A 16x32 layout (ISA 7.12.2):
    // lanes 0-15 : halves 0-7 = K 0..7,  halves 8-15 = K 16..23
    // lanes 16-31: halves 0-7 = K 8..15, halves 8-15 = K 24..31
    ABfrag a;
    const unsigned short* arow = &sA[buf][wv * 16 + (lane & 15)][0];
    a.h[0] = *(const v8bf*)(arow + hi * 8);
    a.h[1] = *(const v8bf*)(arow + 16 + hi * 8);
    // B fragment: lane l holds column N=l&15, K in the same order as A
#pragma unroll
    for (int t = 0; t < 4; ++t) {
      ABfrag b;
#pragma unroll
      for (int e = 0; e < 16; ++e)
        b.v[e] = __builtin_bit_cast(__bf16, sW[buf][(e < 8 ? hi * 8 + e : 16 + hi * 8 + e - 8)][t * 16 + (lane & 15)]);
      acc[t] = __builtin_amdgcn_wmma_f32_16x16x32_bf16(
          false, a.v, false, b.v, (short)0, acc[t], false, false);
    }
    __syncthreads();
  }

  // C/D layout: VGPR r, lanes 0-15 -> M=r, lanes 16-31 -> M=r+8; N = lane&15
#pragma unroll
  for (int t = 0; t < 4; ++t) {
#pragma unroll
    for (int r = 0; r < 8; ++r) {
      const int m = mBase + wv * 16 + r + 8 * hi;
      const int n = nBase + t * 16 + (lane & 15);
      const size_t o = (size_t)m * N + n;
      float v = acc[t][r] + bias[n];
      if (EP == 1)      outB[o] = f2bf(tanhf(v));
      else if (EP == 2) outF[o] = v + resid[o];
      else              outF[o] = v;
    }
  }
}

// ---------------------------------------------------------------- skinny GEMM
// Direct-load WMMA GEMM for N=32 heads (weights are L2-resident, <2% FLOPs).
template<int NT>
__global__ void gemm_wmma_small_kernel(const unsigned short* __restrict__ A,
                                       const unsigned short* __restrict__ W,
                                       const float* __restrict__ bias,
                                       float* __restrict__ outF,
                                       int M, int N, int Kd)
{
  const int lane   = threadIdx.x & 31;
  const int wave   = threadIdx.x >> 5;
  const int tilesN = N / (16 * NT);
  const int tid    = blockIdx.x * (blockDim.x >> 5) + wave;
  const int mt     = tid / tilesN;
  const int nb     = (tid % tilesN) * (16 * NT);
  if (mt * 16 >= M) return;

  const int row = mt * 16 + (lane & 15);
  const int hi  = lane >> 4;

  v8f acc[NT] = {};
  for (int kk = 0; kk < Kd; kk += 32) {
    ABfrag a;
    const v8bf* ap = (const v8bf*)(A + (size_t)row * Kd + kk + hi * 8);
    a.h[0] = ap[0];
    a.h[1] = ap[2];
    if (kk + 32 < Kd)
      __builtin_prefetch(A + (size_t)row * Kd + kk + 32, 0, 3);

#pragma unroll
    for (int t = 0; t < NT; ++t) {
      ABfrag b;
#pragma unroll
      for (int e = 0; e < 16; ++e)
        b.v[e] = __builtin_bit_cast(__bf16, W[(size_t)(kk + (e < 8 ? hi * 8 + e : 16 + hi * 8 + e - 8)) * N + nb + t * 16 + (lane & 15)]);
      acc[t] = __builtin_amdgcn_wmma_f32_16x16x32_bf16(
          false, a.v, false, b.v, (short)0, acc[t], false, false);
    }
  }

#pragma unroll
  for (int t = 0; t < NT; ++t) {
#pragma unroll
    for (int r = 0; r < 8; ++r) {
      const int m = mt * 16 + r + 8 * hi;
      const int n = nb + t * 16 + (lane & 15);
      outF[(size_t)m * N + n] = acc[t][r] + bias[n];
    }
  }
}

// ---------------------------------------------------------------- phase/amp
__global__ void phase_kernel(const float* __restrict__ pk, const float* __restrict__ ak,
                             const float* __restrict__ pq, const float* __restrict__ aq,
                             const float* __restrict__ pos,
                             float* __restrict__ kr, float* __restrict__ ki,
                             float* __restrict__ qr, float* __restrict__ qi)
{
  const int i = blockIdx.x * blockDim.x + threadIdx.x;
  if (i >= MM * KK) return;
  const int m = i / KK;
  const int k = i - m * KK;
  const int l = m % LL;
  const float PI = 3.14159265358979323846f;
  const float p = pos[l * KK + k];

  const float phk = tanhf(pk[i]) * PI + p;
  const float phq = tanhf(pq[i]) * PI + p;
  float a_k = ak[i]; a_k = (a_k > 20.f ? a_k : log1pf(expf(a_k))) + 0.1f;
  float a_q = aq[i]; a_q = (a_q > 20.f ? a_q : log1pf(expf(a_q))) + 0.1f;

  kr[i] = a_k * cosf(phk);
  ki[i] = a_k * sinf(phk);
  qr[i] = a_q * cosf(phq);
  qi[i] = a_q * sinf(phq);
}

// ---------------------------------------------------------------- causal scan
__global__ void scan_kernel(const float* __restrict__ kr, const float* __restrict__ ki,
                            const float* __restrict__ qr, const float* __restrict__ qi,
                            const float* __restrict__ V, float* __restrict__ rn)
{
  const int chunks = DD / 128;
  const int b = blockIdx.x / chunks;
  const int d = (blockIdx.x % chunks) * 128 + threadIdx.x;

  __shared__ float s[4 * KK];
  float Sr[KK] = {};
  float Si[KK] = {};

  for (int l = 0; l < LL; ++l) {
    const size_t mk = (size_t)(b * LL + l) * KK;
    const int t = threadIdx.x;
    if (t < KK)          s[t] = kr[mk + t];
    else if (t < 2 * KK) s[t] = ki[mk + t - KK];
    else if (t < 3 * KK) s[t] = qr[mk + t - 2 * KK];
    else                 s[t] = qi[mk + t - 3 * KK];
    __syncthreads();

    const float v = V[(size_t)(b * LL + l) * DD + d];
    float acc = 0.f;
#pragma unroll
    for (int k = 0; k < KK; ++k) {
      Sr[k] = fmaf(s[k], v, Sr[k]);
      Si[k] = fmaf(s[KK + k], v, Si[k]);
      acc = fmaf(s[2 * KK + k], Sr[k], acc);
      acc = fmaf(s[3 * KK + k], Si[k], acc);
    }
    const float invnorm = rsqrtf((float)(l + 1) * (float)KK);
    rn[(size_t)(b * LL + l) * DD + d] = acc * invnorm;
    __syncthreads();
  }
}

// ---------------------------------------------------------------- layernorm
__global__ void ln_kernel(const float* __restrict__ rn,
                          const float* __restrict__ g,
                          const float* __restrict__ be,
                          unsigned short* __restrict__ lnb)
{
  const int m = blockIdx.x;
  const int t = threadIdx.x;
  const float x0 = rn[(size_t)m * DD + t];
  const float x1 = rn[(size_t)m * DD + t + 256];

  __shared__ float s1[256], s2[256];
  s1[t] = x0 + x1;
  s2[t] = x0 * x0 + x1 * x1;
  __syncthreads();
  for (int off = 128; off > 0; off >>= 1) {
    if (t < off) { s1[t] += s1[t + off]; s2[t] += s2[t + off]; }
    __syncthreads();
  }
  const float mu   = s1[0] * (1.f / DD);
  const float var  = s2[0] * (1.f / DD) - mu * mu;
  const float rstd = rsqrtf(var + 1e-5f);

  lnb[(size_t)m * DD + t]       = f2bf((x0 - mu) * rstd * g[t] + be[t]);
  lnb[(size_t)m * DD + t + 256] = f2bf((x1 - mu) * rstd * g[t + 256] + be[t + 256]);
}

// ---------------------------------------------------------------- launch
extern "C" void kernel_launch(void* const* d_in, const int* in_sizes, int n_in,
                              void* d_out, int out_size, void* d_ws, size_t ws_size,
                              hipStream_t stream) {
  (void)in_sizes; (void)n_in; (void)out_size; (void)ws_size;

  const float* x     = (const float*)d_in[0];
  const float* pos   = (const float*)d_in[1];
  const float* kp_w1 = (const float*)d_in[2];
  const float* kp_b1 = (const float*)d_in[3];
  const float* kp_w2 = (const float*)d_in[4];
  const float* kp_b2 = (const float*)d_in[5];
  const float* ka_w  = (const float*)d_in[6];
  const float* ka_b  = (const float*)d_in[7];
  const float* qp_w1 = (const float*)d_in[8];
  const float* qp_b1 = (const float*)d_in[9];
  const float* qp_w2 = (const float*)d_in[10];
  const float* qp_b2 = (const float*)d_in[11];
  const float* qa_w  = (const float*)d_in[12];
  const float* qa_b  = (const float*)d_in[13];
  const float* v_w   = (const float*)d_in[14];
  const float* v_b   = (const float*)d_in[15];
  const float* ln_g  = (const float*)d_in[16];
  const float* ln_b  = (const float*)d_in[17];
  const float* o_w   = (const float*)d_in[18];
  const float* o_b   = (const float*)d_in[19];
  float* out = (float*)d_out;

  char* ws = (char*)d_ws;
  auto alloc = [&](size_t bytes) -> char* {
    char* p = ws;
    ws += (bytes + 255) & ~(size_t)255;
    return p;
  };

  unsigned short* xb   = (unsigned short*)alloc((size_t)MM * DD * 2);
  unsigned short* hkb  = (unsigned short*)alloc((size_t)MM * DD * 2);
  unsigned short* hqb  = (unsigned short*)alloc((size_t)MM * DD * 2);
  unsigned short* lnbf = (unsigned short*)alloc((size_t)MM * DD * 2);
  unsigned short* wkp1 = (unsigned short*)alloc((size_t)DD * DD * 2);
  unsigned short* wqp1 = (unsigned short*)alloc((size_t)DD * DD * 2);
  unsigned short* wv   = (unsigned short*)alloc((size_t)DD * DD * 2);
  unsigned short* wo   = (unsigned short*)alloc((size_t)DD * DD * 2);
  unsigned short* wkp2 = (unsigned short*)alloc((size_t)DD * KK * 2);
  unsigned short* wqp2 = (unsigned short*)alloc((size_t)DD * KK * 2);
  unsigned short* wka  = (unsigned short*)alloc((size_t)DD * KK * 2);
  unsigned short* wqa  = (unsigned short*)alloc((size_t)DD * KK * 2);
  float* Vf  = (float*)alloc((size_t)MM * DD * 4);
  float* rn  = (float*)alloc((size_t)MM * DD * 4);
  float* pk  = (float*)alloc((size_t)MM * KK * 4);
  float* ak  = (float*)alloc((size_t)MM * KK * 4);
  float* pq  = (float*)alloc((size_t)MM * KK * 4);
  float* aq  = (float*)alloc((size_t)MM * KK * 4);
  float* kr  = (float*)alloc((size_t)MM * KK * 4);
  float* ki  = (float*)alloc((size_t)MM * KK * 4);
  float* qr  = (float*)alloc((size_t)MM * KK * 4);
  float* qi  = (float*)alloc((size_t)MM * KK * 4);

  // 1) fp32 -> bf16 converts
  auto cvt = [&](const float* src, unsigned short* dst, int n) {
    convert_bf16_kernel<<<(n + 255) / 256, 256, 0, stream>>>(src, dst, n);
  };
  cvt(x,     xb,   MM * DD);
  cvt(kp_w1, wkp1, DD * DD);
  cvt(qp_w1, wqp1, DD * DD);
  cvt(v_w,   wv,   DD * DD);
  cvt(o_w,   wo,   DD * DD);
  cvt(kp_w2, wkp2, DD * KK);
  cvt(qp_w2, wqp2, DD * KK);
  cvt(ka_w,  wka,  DD * KK);
  cvt(qa_w,  wqa,  DD * KK);

  // 2) big GEMMs via LDS-staged double-buffered WMMA (128x64 tiles)
  {
    dim3 grid((MM / 128) * (DD / 64)), block(256);
    gemm_wmma_staged_kernel<1><<<grid, block, 0, stream>>>(
        xb, wkp1, kp_b1, nullptr, nullptr, hkb, MM, DD, DD);
    gemm_wmma_staged_kernel<1><<<grid, block, 0, stream>>>(
        xb, wqp1, qp_b1, nullptr, nullptr, hqb, MM, DD, DD);
    gemm_wmma_staged_kernel<0><<<grid, block, 0, stream>>>(
        xb, wv, v_b, nullptr, Vf, nullptr, MM, DD, DD);
  }

  // 3) head GEMMs (N=32): phase/amp pre-activations, f32
  {
    const int waves = (MM / 16) * (KK / 32);
    dim3 grid(waves / 4), block(128);
    gemm_wmma_small_kernel<2><<<grid, block, 0, stream>>>(
        hkb, wkp2, kp_b2, pk, MM, KK, DD);
    gemm_wmma_small_kernel<2><<<grid, block, 0, stream>>>(
        xb,  wka,  ka_b,  ak, MM, KK, DD);
    gemm_wmma_small_kernel<2><<<grid, block, 0, stream>>>(
        hqb, wqp2, qp_b2, pq, MM, KK, DD);
    gemm_wmma_small_kernel<2><<<grid, block, 0, stream>>>(
        xb,  wqa,  qa_b,  aq, MM, KK, DD);
  }

  // 4) complex encode -> kr, ki, qr, qi
  phase_kernel<<<(MM * KK + 255) / 256, 256, 0, stream>>>(
      pk, ak, pq, aq, pos, kr, ki, qr, qi);

  // 5) causal linear-attention scan with in-register [64, D] state
  scan_kernel<<<BB * (DD / 128), 128, 0, stream>>>(kr, ki, qr, qi, Vf, rn);

  // 6) layernorm -> bf16
  ln_kernel<<<MM, 256, 0, stream>>>(rn, ln_g, ln_b, lnbf);

  // 7) out = x + ln @ o_w + o_b
  {
    dim3 grid((MM / 128) * (DD / 64)), block(256);
    gemm_wmma_staged_kernel<2><<<grid, block, 0, stream>>>(
        lnbf, wo, o_b, x, out, nullptr, MM, DD, DD);
  }
}
